// ReducedLissom_57440892617048
// MI455X (gfx1250) — matrix operand order repaired and checked by reading, not compiled, not run on hardware
//
#include <hip/hip_runtime.h>
#include <hip/hip_bf16.h>

// ---------------------------------------------------------------------------
// ReducedLissom on MI455X (gfx1250, wave32).
// Memory-bound GEMV recurrence. Plan:
//  - afferent pass first: v_wmma_f32_16x16x4_f32 on fp32 W_aff (read once,
//    non-temporal so it does not pollute L2).
//  - precompute Wd = bf16(W_exc - W_inh) (non-temporal reads), stored in a
//    WMMA-A-fragment-swizzled layout. Wd is 169.9 MB -> resident in 192 MB L2.
//  - 10 settling steps stream Wd (L2 hits) through v_wmma_f32_16x16x32_bf16.
//  - K split 8 ways into fp32 partials; finalize sums + clamp(x,0,1)
//    (piecewise_sigmoid with theta in [0,1] is exactly a clamp).
// ---------------------------------------------------------------------------

typedef __attribute__((ext_vector_type(16))) __bf16 v16bf;
typedef __attribute__((ext_vector_type(8)))  __bf16 v8bf;
typedef __attribute__((ext_vector_type(8)))  float  v8f;
typedef __attribute__((ext_vector_type(4)))  float  v4f;
typedef __attribute__((ext_vector_type(2)))  float  v2f;

#define NN      9216          // N_SELF == N_IN
#define NB      4             // batch
#define KT32    (NN / 32)     // 288 bf16 k-tiles
#define MT16    (NN / 16)     // 576 row tiles
#define KSPLIT  8
#define KT_PER  (KT32 / KSPLIT)   // 36
#define KSEG    (NN / KSPLIT)     // 1152
#define STEPS   10

// ---------------------------------------------------------------------------
// Precompute Wd = bf16(W_exc - W_inh), pre-swizzled per 16x32 tile so the
// settling kernel's A-fragment is two coalesced b128 loads per lane.
// A-layout (16-bit A 16x32): lane<16 -> row=lane, K0..7 (chunk0), K16..23
// (chunk1); lane>=16 -> row=lane-16, K8..15 (chunk0), K24..31 (chunk1).
// Tile t=(mt*KT32+kt): chunk0 at t*512 + lane*8 elems, chunk1 at +256.
// W_exc / W_inh are read non-temporally (one-shot stream, keep L2 for Wd).
// ---------------------------------------------------------------------------
__global__ __launch_bounds__(256)
void pack_wd_kernel(const float* __restrict__ Wexc,
                    const float* __restrict__ Winh,
                    __bf16* __restrict__ Wd) {
  int wave = blockIdx.x * 8 + (threadIdx.x >> 5);
  int lane = threadIdx.x & 31;
  int mt = wave / KT32;
  int kt = wave - mt * KT32;
  int row = mt * 16 + (lane & 15);
  int c0  = kt * 32 + (lane >> 4) * 8;           // chunk0 column base
  const float* pe = Wexc + (size_t)row * NN + c0;
  const float* pi = Winh + (size_t)row * NN + c0;
  v4f e0 = __builtin_nontemporal_load((const v4f*)(pe));
  v4f e1 = __builtin_nontemporal_load((const v4f*)(pe + 4));
  v4f e2 = __builtin_nontemporal_load((const v4f*)(pe + 16)); // chunk1 cols +16..23
  v4f e3 = __builtin_nontemporal_load((const v4f*)(pe + 20));
  v4f i0 = __builtin_nontemporal_load((const v4f*)(pi));
  v4f i1 = __builtin_nontemporal_load((const v4f*)(pi + 4));
  v4f i2 = __builtin_nontemporal_load((const v4f*)(pi + 16));
  v4f i3 = __builtin_nontemporal_load((const v4f*)(pi + 20));
  v8bf lo, hi;
#pragma unroll
  for (int j = 0; j < 4; ++j) {
    lo[j]     = (__bf16)(e0[j] - i0[j]);
    lo[4 + j] = (__bf16)(e1[j] - i1[j]);
    hi[j]     = (__bf16)(e2[j] - i2[j]);
    hi[4 + j] = (__bf16)(e3[j] - i3[j]);
  }
  size_t tbase = ((size_t)mt * KT32 + kt) * 512;
  *(v8bf*)(Wd + tbase + (size_t)lane * 8)       = lo;   // RT: stay in L2
  *(v8bf*)(Wd + tbase + 256 + (size_t)lane * 8) = hi;
}

// ---------------------------------------------------------------------------
// Afferent pass: partials[ks][b][i] = sum over k-segment of X[b,k]*Waff[i,k]
// using v_wmma_f32_16x16x4_f32 (fp32 A/B, full precision, same HBM traffic).
// A (f32 16x4): lane<16 -> row=lane, K0/K1 in v0/v1; lane>=16 -> K2/K3.
// B (f32 4x16): lane<16 -> col=lane, K0/K1; lane>=16 -> col=lane-16, K2/K3.
// Only D columns 0..3 (batch) are stored; garbage in cols 4..15 is inert.
// ---------------------------------------------------------------------------
__global__ __launch_bounds__(256)
void afferent_wmma_kernel(const float* __restrict__ X,
                          const float* __restrict__ Waff,
                          float* __restrict__ part) {
  int mt   = blockIdx.x;            // 0..575
  int ks   = threadIdx.x >> 5;      // 0..7 (KSPLIT)
  int lane = threadIdx.x & 31;
  int row  = mt * 16 + (lane & 15);
  int bsel = (lane & 15) & 3;       // lanes 4..15 alias batch rows (unused cols)
  int kofs = (lane >> 4) * 2;
  const float* wrow = Waff + (size_t)row * NN + kofs + ks * KSEG;
  const float* xrow = X + (size_t)bsel * NN + kofs + ks * KSEG;
  v8f acc = {};
#pragma unroll 4
  for (int i = 0; i < KSEG / 4; ++i) {          // uniform, compile-time count
    v2f a = __builtin_nontemporal_load((const v2f*)(wrow));
    v2f b = *(const v2f*)(xrow);
    acc = __builtin_amdgcn_wmma_f32_16x16x4_f32(
        false, a, false, b, (short)0, acc, false, false);
    wrow += 4;
    xrow += 4;
  }
  int n = lane & 15;
  if (n < NB) {
    int mo = (lane >> 4) * 8;       // lanes 16..19 hold rows M=8..15
    float* p = part + ((size_t)ks * NB + n) * NN + mt * 16 + mo;
#pragma unroll
    for (int r = 0; r < 8; ++r) p[r] = acc[r];
  }
}

// ---------------------------------------------------------------------------
// Settling step, bf16 path: stream pre-swizzled Wd (L2-resident), act as bf16
// B fragments. B (bf16 32x16): lane<16 -> col=lane, K0..7 then K16..23;
// lane>=16 -> col=lane-16, K8..15 then K24..31 (same order as A) = two b128 loads.
// ---------------------------------------------------------------------------
__global__ __launch_bounds__(256)
void settle_bf16_kernel(const __bf16* __restrict__ Wd,
                        const __bf16* __restrict__ act,
                        float* __restrict__ part) {
  int mt   = blockIdx.x;
  int ks   = threadIdx.x >> 5;
  int lane = threadIdx.x & 31;
  int bsel = (lane & 15) & 3;
  int kofs = (lane >> 4) * 8;
  int kt0  = ks * KT_PER;
  const __bf16* ap = act + (size_t)bsel * NN + kofs + kt0 * 32;
  const __bf16* wt = Wd + ((size_t)mt * KT32 + kt0) * 512 + (size_t)lane * 8;
  union { v16bf v; v8bf h[2]; } A, B;
  v8f acc = {};
#pragma unroll 2
  for (int i = 0; i < KT_PER; ++i) {            // uniform, compile-time count
    __builtin_prefetch(wt + 4 * 512, 0, 1);     // global_prefetch_b8, 4 tiles ahead
    A.h[0] = *(const v8bf*)(wt);
    A.h[1] = *(const v8bf*)(wt + 256);
    B.h[0] = *(const v8bf*)(ap);
    B.h[1] = *(const v8bf*)(ap + 16);
    acc = __builtin_amdgcn_wmma_f32_16x16x32_bf16(
        false, A.v, false, B.v, (short)0, acc, false, false);
    wt += 512;
    ap += 32;
  }
  int n = lane & 15;
  if (n < NB) {
    int mo = (lane >> 4) * 8;
    float* p = part + ((size_t)ks * NB + n) * NN + mt * 16 + mo;
#pragma unroll
    for (int r = 0; r < 8; ++r) p[r] = acc[r];
  }
}

// ---------------------------------------------------------------------------
// Settling step, fp32 fallback (workspace too small for Wd): reads W_exc and
// W_inh directly, subtracts in-register, v_wmma_f32_16x16x4_f32.
// ---------------------------------------------------------------------------
__global__ __launch_bounds__(256)
void settle_f32_kernel(const float* __restrict__ Wexc,
                       const float* __restrict__ Winh,
                       const float* __restrict__ act,
                       float* __restrict__ part) {
  int mt   = blockIdx.x;
  int ks   = threadIdx.x >> 5;
  int lane = threadIdx.x & 31;
  int row  = mt * 16 + (lane & 15);
  int bsel = (lane & 15) & 3;
  int kofs = (lane >> 4) * 2 + ks * KSEG;
  const float* pe = Wexc + (size_t)row * NN + kofs;
  const float* pi = Winh + (size_t)row * NN + kofs;
  const float* xr = act + (size_t)bsel * NN + kofs;
  v8f acc = {};
#pragma unroll 4
  for (int i = 0; i < KSEG / 4; ++i) {
    v2f e = *(const v2f*)(pe);
    v2f w = *(const v2f*)(pi);
    v2f a = e - w;
    v2f b = *(const v2f*)(xr);
    acc = __builtin_amdgcn_wmma_f32_16x16x4_f32(
        false, a, false, b, (short)0, acc, false, false);
    pe += 4; pi += 4; xr += 4;
  }
  int n = lane & 15;
  if (n < NB) {
    int mo = (lane >> 4) * 8;
    float* p = part + ((size_t)ks * NB + n) * NN + mt * 16 + mo;
#pragma unroll
    for (int r = 0; r < 8; ++r) p[r] = acc[r];
  }
}

// ---------------------------------------------------------------------------
// Finalizers: sum K-split partials, clamp(x,0,1) == piecewise_sigmoid.
// ---------------------------------------------------------------------------
__global__ __launch_bounds__(256)
void finalize_aff_kernel(const float* __restrict__ part,
                         float* __restrict__ aff,
                         __bf16* __restrict__ act_bf,
                         float* __restrict__ act_f) {
  int idx = blockIdx.x * 256 + threadIdx.x;     // over NB*NN
  float s = 0.f;
#pragma unroll
  for (int ks = 0; ks < KSPLIT; ++ks) s += part[(size_t)ks * NB * NN + idx];
  s = fminf(fmaxf(s, 0.f), 1.f);
  aff[idx]    = s;
  act_bf[idx] = (__bf16)s;
  act_f[idx]  = s;
}

__global__ __launch_bounds__(256)
void finalize_step_kernel(const float* __restrict__ part,
                          const float* __restrict__ aff,
                          __bf16* __restrict__ act_bf_next,
                          float* __restrict__ act_f_next,
                          float* __restrict__ dout,
                          int write_out) {
  int idx = blockIdx.x * 256 + threadIdx.x;
  float s = aff[idx];
#pragma unroll
  for (int ks = 0; ks < KSPLIT; ++ks) s += part[(size_t)ks * NB * NN + idx];
  s = fminf(fmaxf(s, 0.f), 1.f);
  act_bf_next[idx] = (__bf16)s;
  act_f_next[idx]  = s;
  if (write_out) dout[idx] = s;
}

// ---------------------------------------------------------------------------
extern "C" void kernel_launch(void* const* d_in, const int* in_sizes, int n_in,
                              void* d_out, int out_size, void* d_ws, size_t ws_size,
                              hipStream_t stream) {
  (void)in_sizes; (void)n_in; (void)out_size;
  const float* X    = (const float*)d_in[0];   // cortex_input [4, 9216]
  const float* Waff = (const float*)d_in[1];   // [9216, 9216]
  const float* Wexc = (const float*)d_in[2];
  const float* Winh = (const float*)d_in[3];
  float* out = (float*)d_out;                  // [4, 9216] fp32

  // Workspace carve (256B aligned); small buffers first so fallback fits.
  char* ws = (char*)d_ws;
  size_t off = 0;
  auto carve = [&](size_t bytes) -> size_t {
    size_t o = off;
    off = (off + bytes + 255) & ~(size_t)255;
    return o;
  };
  size_t aff_o     = carve((size_t)NB * NN * sizeof(float));
  size_t actf_o[2] = { carve((size_t)NB * NN * sizeof(float)),
                       carve((size_t)NB * NN * sizeof(float)) };
  size_t actb_o[2] = { carve((size_t)NB * NN * sizeof(__bf16)),
                       carve((size_t)NB * NN * sizeof(__bf16)) };
  size_t part_o    = carve((size_t)KSPLIT * NB * NN * sizeof(float));
  size_t wd_o      = carve((size_t)NN * NN * sizeof(__bf16));
  bool use_wd = (ws_size >= off);              // ~171.6 MB needed for bf16 path

  float*  aff     = (float*)(ws + aff_o);
  float*  actf[2] = { (float*)(ws + actf_o[0]), (float*)(ws + actf_o[1]) };
  __bf16* actb[2] = { (__bf16*)(ws + actb_o[0]), (__bf16*)(ws + actb_o[1]) };
  float*  part    = (float*)(ws + part_o);
  __bf16* Wd      = (__bf16*)(ws + wd_o);

  const int finalize_blocks = (NB * NN) / 256;   // 144

  // Afferent pass FIRST (its 340 MB W_aff stream is non-temporal and happens
  // before Wd is staged, keeping L2 free for Wd residency afterwards).
  afferent_wmma_kernel<<<MT16, 256, 0, stream>>>(X, Waff, part);
  finalize_aff_kernel<<<finalize_blocks, 256, 0, stream>>>(part, aff, actb[0], actf[0]);

  if (use_wd) {
    // Stage Wd into L2 immediately before the settling loop consumes it.
    pack_wd_kernel<<<(MT16 * KT32) / 8, 256, 0, stream>>>(Wexc, Winh, Wd);
  }

  // 10 settling steps (sequential dependence); Wd reads mostly hit L2.
  for (int step = 0; step < STEPS; ++step) {
    int cur = step & 1, nxt = cur ^ 1;
    if (use_wd) {
      settle_bf16_kernel<<<MT16, 256, 0, stream>>>(Wd, actb[cur], part);
    } else {
      settle_f32_kernel<<<MT16, 256, 0, stream>>>(Wexc, Winh, actf[cur], part);
    }
    finalize_step_kernel<<<finalize_blocks, 256, 0, stream>>>(
        part, aff, actb[nxt], actf[nxt], out, step == STEPS - 1 ? 1 : 0);
  }
}
